// MolecularGraphNeuralNetwork_83726092468549
// MI455X (gfx1250) — matrix operand order repaired and checked by reading, not compiled, not run on hardware
//
#include <hip/hip_runtime.h>

// ---------------------------------------------------------------------------
// MGNN fused forward for MI455X (gfx1250).
// One workgroup (256 thr = 8 wave32) per molecule; xs/hs1/hs2/hs3 live in LDS
// across all 3 layers; all GEMMs via v_wmma_f32_16x16x32_bf16 (f32 accum).
// Every WMMA operand is 2x ds_load_b128 (K always LDS-contiguous per lane) and
// every result tile is stored PACKED (one ds_store_b128) by choosing the GEMM
// orientation per matrix:
//   hs1/hs2 : computed transposed (W @ xs^T)   -> packed row-major stores
//   hs3     : computed straight  (xs @ W^T)    -> packed transposed [e][m] store
//   S       : computed transposed (hs2 @ hs1^T)-> float4 adjacency mask +
//                                                  packed staging store
//   O       : computed transposed (hs3^T @ S^T)-> LayerNorm needs only one
//                                                  shfl_xor(16) per statistic,
//                                                  residual I/O is float4/b128
// Only global traffic: embeddings, weights, adjacency stream, output
// -> ~1 GB total, ~45us floor at 23.3 TB/s (memory-bound).
// ---------------------------------------------------------------------------

#define NATOM   256
#define DIM     64
#define NLAYER  3
#define EPSV    1e-6f

typedef __bf16 bf16_t;
typedef __attribute__((ext_vector_type(16))) __bf16 v16bf;
typedef __attribute__((ext_vector_type(8)))  float  v8f;
typedef __attribute__((ext_vector_type(4)))  unsigned int v4u;

union FragAB { v16bf v; v4u q[2]; };
union Pack8  { v4u q; bf16_t b[8]; };

// ---- LDS layout (byte offsets into dynamic shared) ----
#define XF_STRIDE  68      // f32 xs rows: 272B (16B-aligned), 4-bank row rotation
#define HB_STRIDE  72      // bf16 rows: 144B (16B-aligned), 36 banks/row
#define H3T_STRIDE 264     // bf16 transposed hs3 rows: 528B, 16B-aligned
#define WT_STRIDE  72
#define SB_STRIDE  40      // per-wave S staging tile: 80B rows, 16B-aligned
#define WT_MAT     (DIM * WT_STRIDE)
#define OFF_XF 0
#define SZ_XF  (NATOM * XF_STRIDE * 4)
#define OFF_XB (OFF_XF + SZ_XF)
#define SZ_HB  (NATOM * HB_STRIDE * 2)
#define OFF_H1 (OFF_XB + SZ_HB)
#define OFF_H2 (OFF_H1 + SZ_HB)
#define OFF_H3T (OFF_H2 + SZ_HB)
#define SZ_H3T (DIM * H3T_STRIDE * 2)
#define OFF_WT (OFF_H3T + SZ_H3T)
#define SZ_WT  (3 * DIM * WT_STRIDE * 2)
#define OFF_BI (OFF_WT + SZ_WT)
#define SZ_BI  (3 * DIM * 4)
#define OFF_LG (OFF_BI + SZ_BI)
#define OFF_LB (OFF_LG + DIM * 4)
#define OFF_SB (OFF_LB + DIM * 4)
#define SZ_SB  (8 * 16 * SB_STRIDE * 2)
#define SMEM_BYTES (OFF_SB + SZ_SB)    // = 253184 B < 320 KB/WGP

static __device__ __forceinline__ bf16_t f2bf(float f) { return (bf16_t)f; }

static __device__ __forceinline__ v8f wmma_bf16(v16bf a, v16bf b, v8f c) {
  // (neg_a, A, neg_b, B, c_mod, C, reuse_a, reuse_b)
  return __builtin_amdgcn_wmma_f32_16x16x32_bf16(false, a, false, b, (short)0, c,
                                                 false, false);
}

// A fragment (16x32 bf16, MxK): lane = M (lane&15); kh = lane>>4.
// VGPR 0..3: K = kh*8 + {0..7};  VGPR 4..7: K = 16 + kh*8 + {0..7}.
static __device__ __forceinline__ v16bf load_a_frag(const bf16_t* base, int stride,
                                                    int m0, int k0, int lane) {
  FragAB f;
  const bf16_t* p = base + (m0 + (lane & 15)) * stride + k0 + ((lane >> 4) << 3);
  f.q[0] = *(const v4u*)(p);
  f.q[1] = *(const v4u*)(p + 16);
  return f.v;
}

// B fragment (32x16 bf16, KxN) with K contiguous in storage:
// B[k][n] = src[n0+n][k0+k]; VGPR 0..3: K = kh*8 + {0..7}; VGPR 4..7: K = 16 + kh*8 + {0..7}.
static __device__ __forceinline__ v16bf load_b_kcontig(const bf16_t* base, int stride,
                                                       int k0, int n0, int lane) {
  FragAB f;
  const bf16_t* p = base + (n0 + (lane & 15)) * stride + k0 + ((lane >> 4) << 3);
  f.q[0] = *(const v4u*)(p);
  f.q[1] = *(const v4u*)(p + 16);
  return f.v;
}

extern "C" __global__ __launch_bounds__(256, 1)
void mgnn_fused_kernel(const int*   __restrict__ fingerprints,   // [B,N]
                       const float* __restrict__ adj,            // [B,N,N]
                       const float* __restrict__ word_emb,       // [NFP,D]
                       const float* __restrict__ pos_emb,        // [N,D]
                       const float* __restrict__ emb_g,
                       const float* __restrict__ emb_b,
                       const float* __restrict__ W1, const float* __restrict__ b1,
                       const float* __restrict__ W2, const float* __restrict__ b2,
                       const float* __restrict__ W3, const float* __restrict__ b3,
                       const float* __restrict__ ln_g,
                       const float* __restrict__ ln_b,
                       float* __restrict__ out) {
  extern __shared__ char smem[];
  float*  XF  = (float*)(smem + OFF_XF);           // [256][68] f32 xs (residual)
  bf16_t* XB  = (bf16_t*)(smem + OFF_XB);          // [256][72] bf16 xs
  bf16_t* H1  = (bf16_t*)(smem + OFF_H1);          // [256][72] (B operand, K=d)
  bf16_t* H2  = (bf16_t*)(smem + OFF_H2);          // [256][72] (A operand)
  bf16_t* H3T = (bf16_t*)(smem + OFF_H3T);         // [64][264] hs3^T (A operand)
  bf16_t* WT  = (bf16_t*)(smem + OFF_WT);          // 3 x [64][72], e-major (= W)
  float*  BI  = (float*)(smem + OFF_BI);           // 3 x [64]
  float*  LG  = (float*)(smem + OFF_LG);
  float*  LB  = (float*)(smem + OFF_LB);
  bf16_t* SB  = (bf16_t*)(smem + OFF_SB);          // per-wave [16][40] S staging

  const int b    = blockIdx.x;
  const int tid  = threadIdx.x;
  const int lane = tid & 31;
  const int wave = tid >> 5;
  const int nloc = lane & 15;
  const int hh   = lane >> 4;                      // lane half

  // ---------------- embedding + LayerNorm (one row per thread) --------------
  {
    const int n  = tid;
    const int fp = fingerprints[b * NATOM + n];
    const float4* we4 = (const float4*)(word_emb + (size_t)fp * DIM);
    const float4* pe4 = (const float4*)(pos_emb + (size_t)n * DIM);
    float v[DIM];
    float s = 0.f;
#pragma unroll
    for (int i = 0; i < DIM / 4; ++i) {
      float4 a = we4[i], p = pe4[i];
      v[4 * i + 0] = a.x + p.x;  v[4 * i + 1] = a.y + p.y;
      v[4 * i + 2] = a.z + p.z;  v[4 * i + 3] = a.w + p.w;
      s += v[4 * i] + v[4 * i + 1] + v[4 * i + 2] + v[4 * i + 3];
    }
    const float mu = s * (1.f / DIM);
    float q = 0.f;
#pragma unroll
    for (int i = 0; i < DIM; ++i) { float d = v[i] - mu; q += d * d; }
    const float inv = rsqrtf(q * (1.f / DIM) + EPSV);
#pragma unroll
    for (int i = 0; i < DIM; ++i) {
      float y = (v[i] - mu) * inv * emb_g[i] + emb_b[i];
      XF[n * XF_STRIDE + i] = y;
      XB[n * HB_STRIDE + i] = f2bf(y);
    }
  }
  if (tid < DIM) { LG[tid] = ln_g[tid]; LB[tid] = ln_b[tid]; }
  __syncthreads();

  // ------------------------------- layers -----------------------------------
  for (int l = 0; l < NLAYER; ++l) {
    // stage weights: W is [e][d] row-major: K-contiguous as B[k=d][n=e] AND
    // directly usable as A rows-e with K=d.
    for (int idx = tid; idx < DIM * DIM; idx += 256) {
      const int e = idx >> 6, d = idx & 63;
      WT[0 * WT_MAT + e * WT_STRIDE + d] = f2bf(W1[l * 4096 + idx]);
      WT[1 * WT_MAT + e * WT_STRIDE + d] = f2bf(W2[l * 4096 + idx]);
      WT[2 * WT_MAT + e * WT_STRIDE + d] = f2bf(W3[l * 4096 + idx]);
    }
    if (tid < DIM) {
      BI[0 * DIM + tid] = b1[l * DIM + tid];
      BI[1 * DIM + tid] = b2[l * DIM + tid];
      BI[2 * DIM + tid] = b3[l * DIM + tid];
    }
    __syncthreads();

    // ---- hs{1,2,3} = relu(xs @ W^T + bias) : 16 row-tiles over 8 waves ----
    for (int rt = wave; rt < 16; rt += 8) {
      const int m0    = rt << 4;
      const int mcol  = m0 + nloc;
      const int mbase = m0 + (hh << 3);
      // xs as B (K=d contiguous) for the transposed hs1/hs2 tiles:
      const v16bf xb0 = load_b_kcontig(XB, HB_STRIDE, 0,  m0, lane);
      const v16bf xb1 = load_b_kcontig(XB, HB_STRIDE, 32, m0, lane);
      // xs as A (rows m) for the straight hs3 tiles:
      const v16bf xa0 = load_a_frag(XB, HB_STRIDE, m0, 0,  lane);
      const v16bf xa1 = load_a_frag(XB, HB_STRIDE, m0, 32, lane);

      // hs1, hs2 transposed: D^T[e][m] = sum_d W[e][d] xs[m][d]
#pragma unroll
      for (int mat = 0; mat < 2; ++mat) {
        const bf16_t* wt = WT + mat * WT_MAT;
        bf16_t* H = (mat == 0) ? H1 : H2;
#pragma unroll
        for (int t = 0; t < 4; ++t) {
          v8f acc = {0.f, 0.f, 0.f, 0.f, 0.f, 0.f, 0.f, 0.f};
          acc = wmma_bf16(load_a_frag(wt, WT_STRIDE, t * 16, 0,  lane), xb0, acc);
          acc = wmma_bf16(load_a_frag(wt, WT_STRIDE, t * 16, 32, lane), xb1, acc);
          const int e0 = t * 16 + (hh << 3);        // e contiguous along VGPRs
          const float4 bA = *(const float4*)&BI[mat * DIM + e0];
          const float4 bB = *(const float4*)&BI[mat * DIM + e0 + 4];
          Pack8 pk;
          pk.b[0] = f2bf(fmaxf(acc[0] + bA.x, 0.f));
          pk.b[1] = f2bf(fmaxf(acc[1] + bA.y, 0.f));
          pk.b[2] = f2bf(fmaxf(acc[2] + bA.z, 0.f));
          pk.b[3] = f2bf(fmaxf(acc[3] + bA.w, 0.f));
          pk.b[4] = f2bf(fmaxf(acc[4] + bB.x, 0.f));
          pk.b[5] = f2bf(fmaxf(acc[5] + bB.y, 0.f));
          pk.b[6] = f2bf(fmaxf(acc[6] + bB.z, 0.f));
          pk.b[7] = f2bf(fmaxf(acc[7] + bB.w, 0.f));
          *(v4u*)(&H[mcol * HB_STRIDE + e0]) = pk.q;    // packed b128 store
        }
      }
      // hs3 straight: D[m][e]; VGPR axis = consecutive m -> packed H3T[e][m]
      {
        const bf16_t* wt = WT + 2 * WT_MAT;
#pragma unroll
        for (int t = 0; t < 4; ++t) {
          v8f acc = {0.f, 0.f, 0.f, 0.f, 0.f, 0.f, 0.f, 0.f};
          acc = wmma_bf16(xa0, load_b_kcontig(wt, WT_STRIDE, 0,  t * 16, lane), acc);
          acc = wmma_bf16(xa1, load_b_kcontig(wt, WT_STRIDE, 32, t * 16, lane), acc);
          const int   e    = t * 16 + nloc;
          const float bias = BI[2 * DIM + e];
          Pack8 pk;
#pragma unroll
          for (int v = 0; v < 8; ++v)
            pk.b[v] = f2bf(fmaxf(acc[v] + bias, 0.f));
          *(v4u*)(&H3T[e * H3T_STRIDE + mbase]) = pk.q;
        }
      }
    }
    __syncthreads();

    // ---- fused  xs += LN( (hs1 hs2^T * A) @ hs3 ) ----
    for (int rt = wave; rt < 16; rt += 8) {
      const int m0 = rt << 4;
      // S^T[j][m] = sum_d hs2[j][d] hs1[m][d] : B = hs1^T (hoisted)
      const v16bf qb0 = load_b_kcontig(H1, HB_STRIDE, 0,  m0, lane);
      const v16bf qb1 = load_b_kcontig(H1, HB_STRIDE, 32, m0, lane);
      bf16_t* sb = SB + wave * 16 * SB_STRIDE;
      const float* adjb = adj + (size_t)b * NATOM * NATOM;
      v8f O[4];                                     // O^T tiles: rows e, cols m
#pragma unroll
      for (int t = 0; t < 4; ++t) O[t] = {0.f, 0.f, 0.f, 0.f, 0.f, 0.f, 0.f, 0.f};

      for (int jj = 0; jj < NATOM; jj += 32) {
        if (jj + 32 < NATOM)   // stream-ahead hint for next adjacency chunk
          __builtin_prefetch(adjb + (size_t)(m0 + nloc) * NATOM + jj + 32, 0, 0);
#pragma unroll
        for (int half = 0; half < 2; ++half) {
          const int j0 = jj + half * 16;
          v8f S = {0.f, 0.f, 0.f, 0.f, 0.f, 0.f, 0.f, 0.f};
          S = wmma_bf16(load_a_frag(H2, HB_STRIDE, j0, 0,  lane), qb0, S);
          S = wmma_bf16(load_a_frag(H2, HB_STRIDE, j0, 32, lane), qb1, S);
          // lane holds S^T[j0+8*hh+v][m0+nloc]; j contiguous along VGPRs
          const float4* a4 = (const float4*)(adjb +
              (size_t)(m0 + nloc) * NATOM + j0 + (hh << 3));
          const float4 A0 = a4[0], A1 = a4[1];
          Pack8 pk;
          pk.b[0] = f2bf(S[0] * A0.x);  pk.b[1] = f2bf(S[1] * A0.y);
          pk.b[2] = f2bf(S[2] * A0.z);  pk.b[3] = f2bf(S[3] * A0.w);
          pk.b[4] = f2bf(S[4] * A1.x);  pk.b[5] = f2bf(S[5] * A1.y);
          pk.b[6] = f2bf(S[6] * A1.z);  pk.b[7] = f2bf(S[7] * A1.w);
          // SBt[m][j]: masked S tile, j contiguous -> packed b128 store
          *(v4u*)(&sb[nloc * SB_STRIDE + half * 16 + (hh << 3)]) = pk.q;
        }
        // O^T[e][m] += sum_j hs3^T[e][j] S^T[j][m] : A = H3T rows e, B = SBt
        const v16bf sbB = load_b_kcontig(sb, SB_STRIDE, 0, 0, lane);
#pragma unroll
        for (int t = 0; t < 4; ++t)
          O[t] = wmma_bf16(load_a_frag(H3T, H3T_STRIDE, t * 16, jj, lane), sbB, O[t]);
      }

      // LayerNorm along D=64: e runs along tiles/VGPRs within the lane; the
      // partner lane (lane^16) holds the other 32 features of the SAME row m.
      float s = 0.f;
#pragma unroll
      for (int t = 0; t < 4; ++t)
#pragma unroll
        for (int v = 0; v < 8; ++v) s += O[t][v];
      s += __shfl_xor(s, 16, 32);
      const float mu = s * (1.f / DIM);
      float q = 0.f;
#pragma unroll
      for (int t = 0; t < 4; ++t)
#pragma unroll
        for (int v = 0; v < 8; ++v) { float d = O[t][v] - mu; q += d * d; }
      q += __shfl_xor(q, 16, 32);
      const float inv = rsqrtf(q * (1.f / DIM) + EPSV);

      const int m = m0 + nloc;
#pragma unroll
      for (int t = 0; t < 4; ++t) {
        const int e0 = t * 16 + (hh << 3);          // 8 consecutive features
        const float4 g0 = *(const float4*)&LG[e0];
        const float4 g1 = *(const float4*)&LG[e0 + 4];
        const float4 c0 = *(const float4*)&LB[e0];
        const float4 c1 = *(const float4*)&LB[e0 + 4];
        float4 x0 = *(const float4*)&XF[m * XF_STRIDE + e0];
        float4 x1 = *(const float4*)&XF[m * XF_STRIDE + e0 + 4];
        x0.x += (O[t][0] - mu) * inv * g0.x + c0.x;
        x0.y += (O[t][1] - mu) * inv * g0.y + c0.y;
        x0.z += (O[t][2] - mu) * inv * g0.z + c0.z;
        x0.w += (O[t][3] - mu) * inv * g0.w + c0.w;
        x1.x += (O[t][4] - mu) * inv * g1.x + c1.x;
        x1.y += (O[t][5] - mu) * inv * g1.y + c1.y;
        x1.z += (O[t][6] - mu) * inv * g1.z + c1.z;
        x1.w += (O[t][7] - mu) * inv * g1.w + c1.w;
        *(float4*)&XF[m * XF_STRIDE + e0]     = x0;
        *(float4*)&XF[m * XF_STRIDE + e0 + 4] = x1;
        Pack8 pk;
        pk.b[0] = f2bf(x0.x); pk.b[1] = f2bf(x0.y);
        pk.b[2] = f2bf(x0.z); pk.b[3] = f2bf(x0.w);
        pk.b[4] = f2bf(x1.x); pk.b[5] = f2bf(x1.y);
        pk.b[6] = f2bf(x1.z); pk.b[7] = f2bf(x1.w);
        *(v4u*)(&XB[m * HB_STRIDE + e0]) = pk.q;    // packed b128 store
      }
    }
    __syncthreads();
  }

  // ------------------------------ store output ------------------------------
  for (int idx = tid; idx < NATOM * DIM; idx += 256) {
    const int n = idx >> 6, i = idx & 63;
    out[(size_t)b * NATOM * DIM + idx] = XF[n * XF_STRIDE + i];
  }
}

// ---------------------------------------------------------------------------
extern "C" void kernel_launch(void* const* d_in, const int* in_sizes, int n_in,
                              void* d_out, int out_size, void* d_ws, size_t ws_size,
                              hipStream_t stream) {
  (void)in_sizes; (void)n_in; (void)out_size; (void)d_ws; (void)ws_size;
  const int*   fingerprints = (const int*)  d_in[0];
  const float* adjacencies  = (const float*)d_in[1];
  // d_in[2] = molecular_sizes : unused by the reference computation
  const float* word_emb = (const float*)d_in[3];
  const float* pos_emb  = (const float*)d_in[4];
  const float* emb_g    = (const float*)d_in[5];
  const float* emb_b    = (const float*)d_in[6];
  const float* W1 = (const float*)d_in[7];
  const float* b1 = (const float*)d_in[8];
  const float* W2 = (const float*)d_in[9];
  const float* b2 = (const float*)d_in[10];
  const float* W3 = (const float*)d_in[11];
  const float* b3 = (const float*)d_in[12];
  const float* ln_g = (const float*)d_in[13];
  const float* ln_b = (const float*)d_in[14];
  float* out = (float*)d_out;

  hipFuncSetAttribute((const void*)mgnn_fused_kernel,
                      hipFuncAttributeMaxDynamicSharedMemorySize, SMEM_BYTES);
  mgnn_fused_kernel<<<dim3(1024), dim3(256), SMEM_BYTES, stream>>>(
      fingerprints, adjacencies, word_emb, pos_emb, emb_g, emb_b,
      W1, b1, W2, b2, W3, b3, ln_g, ln_b, out);
}
